// Model_30099130810450
// MI455X (gfx1250) — matrix operand order repaired and checked by reading, not compiled, not run on hardware
//
#include <hip/hip_runtime.h>
#include <hip/hip_bf16.h>
#include <math.h>

#ifndef __has_builtin
#define __has_builtin(x) 0
#endif

// ---------------------------------------------------------------------------
// CDNA5 / gfx1250 types for WMMA + async copies
// ---------------------------------------------------------------------------
typedef __attribute__((ext_vector_type(8)))  __bf16 v8bf;
typedef __attribute__((ext_vector_type(16))) __bf16 v16bf;
typedef __attribute__((ext_vector_type(8)))  float  v8f;
typedef int v4i __attribute__((__vector_size__(16)));

typedef __attribute__((address_space(1))) v4i gas_v4i;   // global
typedef __attribute__((address_space(3))) v4i las_v4i;   // LDS

// ---------------------------------------------------------------------------
// CDNA5 async global->LDS (GLOBAL_LOAD_ASYNC_TO_LDS_B128, ASYNCcnt-tracked)
// ---------------------------------------------------------------------------
__device__ __forceinline__ void async_b128(const __bf16* g, __bf16* l) {
#if __has_builtin(__builtin_amdgcn_global_load_async_to_lds_b128)
    __builtin_amdgcn_global_load_async_to_lds_b128(
        (gas_v4i*)(unsigned long long)(const void*)g,
        (las_v4i*)(unsigned)(unsigned long long)(void*)l,
        0, 0);
#else
    unsigned lo = (unsigned)(unsigned long long)(void*)l;
    unsigned long long ga = (unsigned long long)(const void*)g;
    asm volatile("global_load_async_to_lds_b128 %0, %1, off"
                 :: "v"(lo), "v"(ga) : "memory");
#endif
}

__device__ __forceinline__ void wait_async_0() {
#if __has_builtin(__builtin_amdgcn_s_wait_asynccnt)
    __builtin_amdgcn_s_wait_asynccnt(0);
#else
    asm volatile("s_wait_asynccnt 0" ::: "memory");
#endif
}

__device__ __forceinline__ void wait_async_2() {
#if __has_builtin(__builtin_amdgcn_s_wait_asynccnt)
    __builtin_amdgcn_s_wait_asynccnt(2);
#else
    asm volatile("s_wait_asynccnt 2" ::: "memory");
#endif
}

// ---------------------------------------------------------------------------
// Helpers
// ---------------------------------------------------------------------------
__device__ __forceinline__ float mish_f(float x) {
    float sp = (x > 20.0f) ? x : log1pf(expf(x));
    return x * tanhf(sp);
}

// ---------------------------------------------------------------------------
// Weight convert+transpose: W fp32 [K,N] row-major -> WT bf16 [Npad,K] row-major
// ---------------------------------------------------------------------------
__global__ void wtrans_kernel(const float* __restrict__ W, __bf16* __restrict__ WT,
                              int K, int N, int Npad) {
    int i = blockIdx.x * blockDim.x + threadIdx.x;
    if (i >= Npad * K) return;
    int n = i / K;
    int k = i - n * K;
    float v = (n < N) ? W[(size_t)k * N + n] : 0.0f;
    WT[i] = (__bf16)v;
}

// fp32 -> bf16 (optionally out = bf16(a + b))
__global__ void to_bf16_kernel(const float* __restrict__ a, const float* __restrict__ b,
                               __bf16* __restrict__ out, int n) {
    int i = blockIdx.x * blockDim.x + threadIdx.x;
    if (i >= n) return;
    float v = a[i];
    if (b) v += b[i];
    out[i] = (__bf16)v;
}

// ---------------------------------------------------------------------------
// LayerNorm over last dim D; one block (256 thr) per row; fp32 + bf16 outputs
// ---------------------------------------------------------------------------
__global__ void layernorm_kernel(const float* __restrict__ x,
                                 const float* __restrict__ g,
                                 const float* __restrict__ bta,
                                 float* __restrict__ y,
                                 __bf16* __restrict__ ybf,
                                 int D) {
    const int row = blockIdx.x;
    const float* xr = x + (size_t)row * D;
    float s = 0.f, sq = 0.f;
    for (int d = threadIdx.x; d < D; d += blockDim.x) {
        float v = xr[d];
        s += v; sq += v * v;
    }
    __shared__ float shs[256];
    __shared__ float shq[256];
    shs[threadIdx.x] = s;
    shq[threadIdx.x] = sq;
    __syncthreads();
    for (int st = 128; st > 0; st >>= 1) {
        if ((int)threadIdx.x < st) {
            shs[threadIdx.x] += shs[threadIdx.x + st];
            shq[threadIdx.x] += shq[threadIdx.x + st];
        }
        __syncthreads();
    }
    float mean = shs[0] / (float)D;
    float var  = shq[0] / (float)D - mean * mean;
    float inv  = rsqrtf(var + 1e-7f);
    float* yr = y + (size_t)row * D;
    __bf16* yb = ybf ? (ybf + (size_t)row * D) : nullptr;
    for (int d = threadIdx.x; d < D; d += blockDim.x) {
        float v = (xr[d] - mean) * inv * g[d] + bta[d];
        yr[d] = v;
        if (yb) yb[d] = (__bf16)v;
    }
}

// ---------------------------------------------------------------------------
// WMMA GEMM with async-LDS double-buffered B panel.
//   C[M,N] = A[M,K](bf16, ld=K) @ BT[Npad,K](bf16, ld=K)^T
//            + bias[N] (+ resid[M,ldC]) (+ Mish)
// Block: 256 thr = 8 waves; tile M=128 x N=64; K chunked by 64, 2 LDS buffers
// of 64 cols x 64 k bf16 (8 KB each) filled with GLOBAL_LOAD_ASYNC_TO_LDS_B128.
// All 8 waves share the B panel (8x VMEM traffic reduction vs direct loads).
// Fragment layouts used here (wave32; A and B share contiguous k order):
//   A 16x32 bf16: lane row = lane&15; elems 0..15 -> K = half*16+{0..15}
//                 (half = lane>>4)
//   B 32x16 bf16: lane col = lane&15; elems 0..15 -> K = half*16+{0..15}
//   C/D 16x16 f32: vgpr r: M = r + 8*half, N = lane&15
// ---------------------------------------------------------------------------
template <int ACT>
__launch_bounds__(256)
__global__ void wmma_gemm_kernel(const __bf16* __restrict__ A,
                                 const __bf16* __restrict__ BT,
                                 const float* __restrict__ bias,
                                 const float* __restrict__ resid,
                                 float* __restrict__ C,
                                 int M, int N, int K, int ldC) {
    __shared__ __bf16 Bs[2][64][64];

    const int tid  = threadIdx.x;
    const int lane = tid & 31;
    const int wave = tid >> 5;
    const int half = lane >> 4;   // 0 or 1
    const int l16  = lane & 15;

    const int m0 = blockIdx.x * 128 + wave * 16;
    const int n0 = blockIdx.y * 64;

    // cooperative B-panel loader mapping: 4 threads per column, 16 elems each
    const int cl  = tid >> 2;     // 0..63 local column
    const int seg = tid & 3;      // 0..3 k-segment of 16 elems
    const __bf16* bsrc = BT + (size_t)(n0 + cl) * K + seg * 16;
    __bf16* bdst0 = &Bs[0][cl][seg * 16];
    __bf16* bdst1 = &Bs[1][cl][seg * 16];

    int arow = m0 + l16;
    if (arow >= M) arow = M - 1;          // clamp for ragged M; stores guarded
    const __bf16* Arow = A + (size_t)arow * K + half * 16;

    v8f acc[4] = {};
    const int nc = K >> 6;                // #chunks of 64 (K is a multiple of 64)

    // prologue: chunk 0 -> buffer 0 (2 async instructions per wave)
    async_b128(bsrc, bdst0);
    async_b128(bsrc + 8, bdst0 + 8);

    for (int c = 0; c < nc; ++c) {
        const int k0  = c << 6;
        const int buf = c & 1;

        if (c + 1 < nc) {                 // prefetch next chunk into other buffer
            const __bf16* s = bsrc + ((c + 1) << 6);
            __bf16* d = buf ? bdst0 : bdst1;
            async_b128(s, d);
            async_b128(s + 8, d + 8);
            wait_async_2();               // chunk c landed (in-order completion)
        } else {
            wait_async_0();
        }
        __syncthreads();                  // panel visible to all 8 waves

#pragma unroll
        for (int ks = 0; ks < 2; ++ks) {  // two 32-wide K steps per chunk
            v8bf lo = *(const v8bf*)(Arow + k0 + ks * 32);
            v8bf hi = *(const v8bf*)(Arow + k0 + ks * 32 + 8);
            v16bf afrag;
#pragma unroll
            for (int i = 0; i < 8; ++i) { afrag[i] = lo[i]; afrag[i + 8] = hi[i]; }

#pragma unroll
            for (int nt = 0; nt < 4; ++nt) {
                const v16bf bfrag =
                    *(const v16bf*)&Bs[buf][nt * 16 + l16][ks * 32 + half * 16];
                acc[nt] = __builtin_amdgcn_wmma_f32_16x16x32_bf16(
                    false, afrag, false, bfrag, (short)0, acc[nt], false, false);
            }
        }
        __syncthreads();                  // protect buffer before next prefetch
    }

#pragma unroll
    for (int nt = 0; nt < 4; ++nt) {
        const int n = n0 + nt * 16 + l16;
#pragma unroll
        for (int r = 0; r < 8; ++r) {
            const int m = m0 + r + half * 8;
            if (m < M && n < N) {
                float v = acc[nt][r];
                if (bias)  v += bias[n];
                if (resid) v += resid[(size_t)m * ldC + n];
                if (ACT == 1) v = mish_f(v);
                C[(size_t)m * ldC + n] = v;
            }
        }
    }
}

// ---------------------------------------------------------------------------
// Disentangled attention, one block per (batch, head), one thread per query row.
// scores[i][j] = (q_i.k_j + q_i.posK[p] + k_j.posQ[p]) * 1/sqrt(3*dh),
// p = clip(i - j + 50, 0, 99). Softmax over j, then ctx = probs @ v.
// ---------------------------------------------------------------------------
template <int DH>
__global__ void attention_kernel(const float* __restrict__ Q,
                                 const float* __restrict__ Km,
                                 const float* __restrict__ V,
                                 const float* __restrict__ PK,
                                 const float* __restrict__ PQ,
                                 float* __restrict__ CTX) {
    constexpr int L = 50;
    constexpr int NH = 16;
    constexpr int D = NH * DH;

    const int bid = blockIdx.x;
    const int b = bid >> 4;
    const int h = bid & 15;
    const int i = threadIdx.x;

    __shared__ float sc[L][L + 1];

    if (i >= L) return;

    const float scale = rsqrtf((float)(3 * DH));
    const size_t base = (size_t)(b * L) * D + (size_t)h * DH;

    float q[DH];
    const float* qi = Q + base + (size_t)i * D;
#pragma unroll
    for (int d = 0; d < DH; ++d) q[d] = qi[d];

    for (int j = 0; j < L; ++j) {
        const float* kj = Km + base + (size_t)j * D;
        int p = i - j + 50;
        p = (p < 0) ? 0 : ((p > 99) ? 99 : p);
        const float* pk = PK + (size_t)p * D + (size_t)h * DH;
        const float* pq = PQ + (size_t)p * D + (size_t)h * DH;
        float a0 = 0.f, a1 = 0.f, a2 = 0.f;
#pragma unroll
        for (int d = 0; d < DH; ++d) {
            float kv = kj[d];
            a0 += q[d] * kv;      // content -> content
            a1 += q[d] * pk[d];   // content -> position (c2p)
            a2 += kv   * pq[d];   // position -> content (p2c)
        }
        sc[i][j] = (a0 + a1 + a2) * scale;
    }

    float mx = sc[i][0];
    for (int j = 1; j < L; ++j) mx = fmaxf(mx, sc[i][j]);
    float sum = 0.f;
    for (int j = 0; j < L; ++j) {
        float e = expf(sc[i][j] - mx);
        sc[i][j] = e;
        sum += e;
    }
    const float inv = 1.0f / sum;

    float acc[DH] = {};
    for (int j = 0; j < L; ++j) {
        const float pr = sc[i][j] * inv;
        const float* vj = V + base + (size_t)j * D;
#pragma unroll
        for (int d = 0; d < DH; ++d) acc[d] += pr * vj[d];
    }
    float* out = CTX + base + (size_t)i * D;
#pragma unroll
    for (int d = 0; d < DH; ++d) out[d] = acc[d];
}

// ---------------------------------------------------------------------------
// Mean pool over L
// ---------------------------------------------------------------------------
__global__ void mean_pool_kernel(const float* __restrict__ h, float* __restrict__ pooled,
                                 int Bsz, int L, int D) {
    int i = blockIdx.x * blockDim.x + threadIdx.x;
    if (i >= Bsz * D) return;
    int b = i / D;
    int d = i - b * D;
    float s = 0.f;
    for (int l = 0; l < L; ++l) s += h[((size_t)b * L + l) * D + d];
    pooled[i] = s / (float)L;
}

// ---------------------------------------------------------------------------
// Host orchestration
// ---------------------------------------------------------------------------
extern "C" void kernel_launch(void* const* d_in, const int* in_sizes, int n_in,
                              void* d_out, int out_size, void* d_ws, size_t ws_size,
                              hipStream_t stream) {
    (void)in_sizes; (void)n_in; (void)out_size; (void)ws_size;

    const int B = 256, L = 50, NH = 16;
    const int M = B * L;                 // 12800 tokens
    const int Dmax = 768;
    const int Din_[3]  = {512, 768, 768};
    const int Dout_[3] = {768, 768, 512};

    // Input layout: x, then per layer 19 tensors in dict insertion order, then Wcls, bcls
    const float* x = (const float*)d_in[0];
    auto LP = [&](int l, int idx) -> const float* {
        return (const float*)d_in[1 + 19 * l + idx];
    };
    // per-layer indices: Wq0 bq1 Wk2 bk3 Wv4 bv5 Wso6 bso7 Wi8 bi9 Wout10 bout11
    //                    rel12 rel_g13 rel_b14 ln1_g15 ln1_b16 ln2_g17 ln2_b18
    const float* Wcls = (const float*)d_in[1 + 19 * 3];
    const float* bcls = (const float*)d_in[1 + 19 * 3 + 1];

    // ---- workspace carve-up -------------------------------------------------
    size_t off = 0;
    auto take = [&](size_t bytes) -> char* {
        char* p = (char*)d_ws + off;
        off += (bytes + 255) & ~(size_t)255;
        return p;
    };
    const size_t actB = (size_t)M * Dmax * sizeof(float);
    float* Qb   = (float*)take(actB);
    float* Kb   = (float*)take(actB);
    float* Vb   = (float*)take(actB);
    float* CTXb = (float*)take(actB);
    float* HAb  = (float*)take(actB);
    float* HBb  = (float*)take(actB);
    __bf16* XB  = (__bf16*)take((size_t)M * Dmax * sizeof(__bf16));
    float* relLN   = (float*)take((size_t)112 * Dmax * sizeof(float));
    __bf16* relLNb = (__bf16*)take((size_t)112 * Dmax * sizeof(__bf16));
    float* posK = (float*)take((size_t)112 * Dmax * sizeof(float));
    float* posQ = (float*)take((size_t)112 * Dmax * sizeof(float));
    float* pooled = (float*)take((size_t)B * 512 * sizeof(float));

    __bf16 *WTq[3], *WTk[3], *WTv[3], *WTso[3], *WTi[3], *WTout[3], *WTcls;
    for (int l = 0; l < 3; ++l) {
        size_t sq = (size_t)Din_[l] * Din_[l] * sizeof(__bf16);
        WTq[l]  = (__bf16*)take(sq);
        WTk[l]  = (__bf16*)take(sq);
        WTv[l]  = (__bf16*)take(sq);
        WTso[l] = (__bf16*)take(sq);
        WTi[l]  = (__bf16*)take(sq);
        WTout[l] = (__bf16*)take((size_t)Dout_[l] * Din_[l] * sizeof(__bf16));
    }
    WTcls = (__bf16*)take((size_t)256 * 512 * sizeof(__bf16));

    // ---- build bf16 transposed weights -------------------------------------
    auto buildWT = [&](const float* W, __bf16* WT, int Kd, int Nd, int Npad) {
        int n = Npad * Kd;
        hipLaunchKernelGGL(wtrans_kernel, dim3((n + 255) / 256), dim3(256), 0, stream,
                           W, WT, Kd, Nd, Npad);
    };
    for (int l = 0; l < 3; ++l) {
        int Din = Din_[l], Dout = Dout_[l];
        buildWT(LP(l, 0),  WTq[l],  Din, Din, Din);
        buildWT(LP(l, 2),  WTk[l],  Din, Din, Din);
        buildWT(LP(l, 4),  WTv[l],  Din, Din, Din);
        buildWT(LP(l, 6),  WTso[l], Din, Din, Din);
        buildWT(LP(l, 8),  WTi[l],  Din, Din, Din);
        buildWT(LP(l, 10), WTout[l], Din, Dout, Dout);
    }
    buildWT(Wcls, WTcls, 512, 250, 256);

    auto gemm = [&](const __bf16* A, const __bf16* BT, const float* bias,
                    const float* resid, float* C, int Mm, int Nn, int Kk, int ldC,
                    int act) {
        dim3 grid((Mm + 127) / 128, (Nn + 63) / 64);
        if (act)
            hipLaunchKernelGGL((wmma_gemm_kernel<1>), grid, dim3(256), 0, stream,
                               A, BT, bias, resid, C, Mm, Nn, Kk, ldC);
        else
            hipLaunchKernelGGL((wmma_gemm_kernel<0>), grid, dim3(256), 0, stream,
                               A, BT, bias, resid, C, Mm, Nn, Kk, ldC);
    };
    auto cvt = [&](const float* a, const float* b, __bf16* o, int n) {
        hipLaunchKernelGGL(to_bf16_kernel, dim3((n + 255) / 256), dim3(256), 0, stream,
                           a, b, o, n);
    };
    auto lnorm = [&](const float* xx, const float* g, const float* bb, float* y,
                     __bf16* yb, int R, int D) {
        hipLaunchKernelGGL(layernorm_kernel, dim3(R), dim3(256), 0, stream,
                           xx, g, bb, y, yb, D);
    };

    // ---- encoder layers -----------------------------------------------------
    for (int l = 0; l < 3; ++l) {
        const int Din = Din_[l], Dout = Dout_[l];
        const float* Hin = (l == 0) ? x : ((l == 1) ? HAb : HBb);
        float* Hout = (l == 1) ? HBb : HAb;

        if (l == 0) cvt(Hin, nullptr, XB, M * Din);   // later layers: LN wrote XB

        // relative-position embedding: LN then K/Q projections (M=100)
        lnorm(LP(l, 12), LP(l, 13), LP(l, 14), relLN, relLNb, 2 * 50, Din);
        gemm(relLNb, WTk[l], LP(l, 3), nullptr, posK, 100, Din, Din, Din, 0);
        gemm(relLNb, WTq[l], LP(l, 1), nullptr, posQ, 100, Din, Din, Din, 0);

        // QKV projections
        gemm(XB, WTq[l], LP(l, 1), nullptr, Qb, M, Din, Din, Din, 0);
        gemm(XB, WTk[l], LP(l, 3), nullptr, Kb, M, Din, Din, Din, 0);
        gemm(XB, WTv[l], LP(l, 5), nullptr, Vb, M, Din, Din, Din, 0);

        // disentangled attention
        if (Din == 512)
            hipLaunchKernelGGL((attention_kernel<32>), dim3(B * NH), dim3(64), 0, stream,
                               Qb, Kb, Vb, posK, posQ, CTXb);
        else
            hipLaunchKernelGGL((attention_kernel<48>), dim3(B * NH), dim3(64), 0, stream,
                               Qb, Kb, Vb, posK, posQ, CTXb);

        // self-output dense + residual, then LN1  (PRE reuses Qb, ATT reuses Kb)
        cvt(CTXb, nullptr, XB, M * Din);
        gemm(XB, WTso[l], LP(l, 7), Hin, Qb, M, Din, Din, Din, 0);
        lnorm(Qb, LP(l, 15), LP(l, 16), Kb, XB, M, Din);

        // intermediate dense with Mish (INT reuses Vb)
        gemm(XB, WTi[l], LP(l, 9), nullptr, Vb, M, Din, Din, Din, 1);

        // modified output: dense(inter + attn_out) -> LN2
        cvt(Vb, Kb, XB, M * Din);                       // XB = bf16(inter + attn_out)
        gemm(XB, WTout[l], LP(l, 11), nullptr, CTXb, M, Dout, Din, Dout, 0);
        lnorm(CTXb, LP(l, 17), LP(l, 18), Hout, XB, M, Dout);  // XB ready for next layer
    }

    // ---- pool + classifier --------------------------------------------------
    {
        int n = B * 512;
        hipLaunchKernelGGL(mean_pool_kernel, dim3((n + 255) / 256), dim3(256), 0, stream,
                           HAb, pooled, B, L, 512);
        cvt(pooled, nullptr, XB, n);
        gemm(XB, WTcls, bcls, nullptr, (float*)d_out, B, 250, 512, 250, 0);
    }
}
